// Head_50878182588521
// MI455X (gfx1250) — matrix operand order repaired and checked by reading, not compiled, not run on hardware
//
#include <hip/hip_runtime.h>
#include <hip/hip_bf16.h>

// ---------------------------------------------------------------------------
// Types matching gfx1250 WMMA builtin signatures
// ---------------------------------------------------------------------------
typedef __attribute__((ext_vector_type(16))) __bf16 v16bf;
typedef __attribute__((ext_vector_type(8)))  __bf16 v8bf;
typedef __attribute__((ext_vector_type(8)))  float  v8f;
typedef __attribute__((ext_vector_type(4)))  float  v4f;

#define BATCH 8
#define SEQ   2048
#define EMB   1024
#define HD    64

static __device__ __forceinline__ v16bf bfcat(v8bf a, v8bf b) {
    return __builtin_shufflevector(a, b, 0,1,2,3,4,5,6,7,8,9,10,11,12,13,14,15);
}

static __device__ __forceinline__ v8f wmma_bf16(v16bf a, v16bf b, v8f c) {
    // D = A(16x32 bf16) * B(32x16 bf16) + C(16x16 f32)
    return __builtin_amdgcn_wmma_f32_16x16x32_bf16(
        /*neg_a=*/false, a, /*neg_b=*/false, b,
        /*c_mod=*/(short)0, c, /*reuse_a=*/false, /*reuse_b=*/false);
}

// B-fragment (32x16, 16-bit): lane l holds column n=l%16, elements e -> K = e + 16*(l/16).
// One contiguous 16-element run => two 16B loads (global_load_b128 or ds_load_b128).
static __device__ __forceinline__ v16bf load_bfrag(const __bf16* p) {
    v8bf a = *(const v8bf*)p;
    v8bf b = *(const v8bf*)(p + 8);
    return bfcat(a, b);
}

// Async global->LDS copy of 32 bytes (two b128), per lane.
// TDM inst_offset is added to BOTH the LDS and the global address.
static __device__ __forceinline__ void async_copy32(const void* gptr, const void* lptr) {
    unsigned           l  = (unsigned)(unsigned long long)lptr;   // LDS byte offset (addr[31:0])
    unsigned long long ga = (unsigned long long)gptr;
    asm volatile("global_load_async_to_lds_b128 %0, %1, off"           :: "v"(l), "v"(ga) : "memory");
    asm volatile("global_load_async_to_lds_b128 %0, %1, off offset:16" :: "v"(l), "v"(ga) : "memory");
}

// ---------------------------------------------------------------------------
// Kernel 0: convert + transpose weights fp32 [1024][64] -> bf16 Wt[3][64][1024]
// ---------------------------------------------------------------------------
__global__ __launch_bounds__(256) void k_convw(const float* __restrict__ Wq,
                                               const float* __restrict__ Wk,
                                               const float* __restrict__ Wv,
                                               __bf16* __restrict__ Wt) {
    int tid = blockIdx.x * blockDim.x + threadIdx.x;
    if (tid >= 3 * HD * EMB) return;
    int w   = tid >> 16;
    int rem = tid & 65535;
    int n   = rem >> 10;
    int k   = rem & 1023;
    const float* src = (w == 0) ? Wq : ((w == 1) ? Wk : Wv);
    Wt[tid] = (__bf16)src[k * HD + n];
}

// ---------------------------------------------------------------------------
// Kernel 1: fused QKV projection. One wave per 16-row tile of x (flat B*T).
//   Q,K stored row-major bf16 [B*T][64];  V stored transposed bf16 [B][64][T].
// ---------------------------------------------------------------------------
__global__ __launch_bounds__(32) void k_proj(const float* __restrict__ x,
                                             const __bf16* __restrict__ Wt,
                                             __bf16* __restrict__ Qo,
                                             __bf16* __restrict__ Ko,
                                             __bf16* __restrict__ Vt) {
    const int lane = threadIdx.x;
    const int half = lane >> 4;
    const int lm   = lane & 15;
    const int rt   = blockIdx.x;
    const int row  = rt * 16 + lm;

    v8f accq[4], acck[4], accv[4];
#pragma unroll
    for (int nt = 0; nt < 4; ++nt) { accq[nt] = (v8f){}; acck[nt] = (v8f){}; accv[nt] = (v8f){}; }

    for (int kb = 0; kb < EMB; kb += 32) {
        // A fragment from fp32 x: element e -> k = kb + 16*half + e
        const float* xp = x + (size_t)row * EMB + kb + half * 16;
        v4f f0 = *(const v4f*)(xp);
        v4f f1 = *(const v4f*)(xp + 4);
        v4f f2 = *(const v4f*)(xp + 8);
        v4f f3 = *(const v4f*)(xp + 12);
        v16bf A;
#pragma unroll
        for (int i = 0; i < 4; ++i) {
            A[i]      = (__bf16)f0[i];
            A[4 + i]  = (__bf16)f1[i];
            A[8 + i]  = (__bf16)f2[i];
            A[12 + i] = (__bf16)f3[i];
        }
#pragma unroll
        for (int nt = 0; nt < 4; ++nt) {
            const int n = nt * 16 + lm;
            const size_t roff = (size_t)n * EMB + kb + half * 16;
            accq[nt] = wmma_bf16(A, load_bfrag(Wt + 0 * HD * EMB + roff), accq[nt]);
            acck[nt] = wmma_bf16(A, load_bfrag(Wt + 1 * HD * EMB + roff), acck[nt]);
            accv[nt] = wmma_bf16(A, load_bfrag(Wt + 2 * HD * EMB + roff), accv[nt]);
        }
    }

    const int b     = (rt * 16) >> 11;
    const int tloc0 = (rt * 16) & 2047;
#pragma unroll
    for (int nt = 0; nt < 4; ++nt) {
        const int n = nt * 16 + lm;
#pragma unroll
        for (int r = 0; r < 8; ++r) {
            const int m = r + half * 8;
            const size_t gt = (size_t)rt * 16 + m;
            Qo[gt * HD + n] = (__bf16)accq[nt][r];
            Ko[gt * HD + n] = (__bf16)acck[nt][r];
            Vt[((size_t)b * HD + n) * SEQ + (tloc0 + m)] = (__bf16)accv[nt][r];
        }
    }
}

// ---------------------------------------------------------------------------
// Kernel 2: flash attention, 4 waves/workgroup (64 queries), 32-key blocks.
// K/V tiles double-buffered in LDS via global_load_async_to_lds_b128.
// ---------------------------------------------------------------------------
__global__ __launch_bounds__(128) void k_attn(const __bf16* __restrict__ Q,
                                              const __bf16* __restrict__ K,
                                              const __bf16* __restrict__ Vt,
                                              const int* __restrict__ pad,
                                              float* __restrict__ out) {
    // [buf][ K tile: 32 keys x 64 hd | V tile: 64 hd x 32 keys ] = 8KB per buffer
    __shared__ __bf16 KV[2][4096];
    __shared__ __bf16 Pl[4][16][32];   // per-wave P tile, C-layout -> A-layout

    const int tid  = threadIdx.x;
    const int wid  = tid >> 5;
    const int lane = tid & 31;
    const int half = lane >> 4;
    const int lm   = lane & 15;
    const int b    = blockIdx.x >> 5;      // 32 query groups per batch
    const int qg   = blockIdx.x & 31;
    const int qt   = qg * 4 + wid;         // this wave's 16-query tile

    // Uniform block count across the workgroup (covers qt = qg*4+3).
    const int nkb = 2 * qg + 2;

    const __bf16* Kb = K  + (size_t)b * SEQ * HD;
    const __bf16* Vb = Vt + (size_t)b * HD * SEQ;

    // Stage one 32-key block into KV[buf]: each of 128 threads copies 32B of K + 32B of V.
    auto issue = [&](int buf, int kb) {
        const int k0 = kb * 32;
        // K block: contiguous 4KB at Kb + k0*64
        async_copy32((const char*)(Kb + (size_t)k0 * HD) + tid * 32,
                     (const char*)&KV[buf][0] + tid * 32);
        // V block: 64 rows (hd) x 64B, global row stride 4KB
        const int n = tid >> 1, h = tid & 1;
        async_copy32((const char*)(Vb + (size_t)n * SEQ + k0) + h * 32,
                     (const char*)&KV[buf][2048] + n * 64 + h * 32);
    };

    // Q A-fragments (head dim 0..31 / 32..63)
    const __bf16* qp = Q + ((size_t)(b * SEQ + qt * 16 + lm)) * HD;
    v16bf A0 = bfcat(*(const v8bf*)(qp + half * 16),      *(const v8bf*)(qp + half * 16 + 8));
    v16bf A1 = bfcat(*(const v8bf*)(qp + 32 + half * 16), *(const v8bf*)(qp + 32 + half * 16 + 8));

    v8f o0 = (v8f){}, o1 = (v8f){}, o2 = (v8f){}, o3 = (v8f){};
    float mrow[8], lrow[8];
#pragma unroll
    for (int r = 0; r < 8; ++r) { mrow[r] = -1e30f; lrow[r] = 0.0f; }

    const float SCL = 0.045084220027780106f;   // (1/sqrt(C)) * log2(e)

    issue(0, 0);   // prologue

    for (int kb = 0; kb < nkb; ++kb) {
        const int  cur  = kb & 1;
        const bool more = (kb + 1) < nkb;
        if (more) {
            issue(cur ^ 1, kb + 1);                               // prefetch next block
            asm volatile("s_wait_asynccnt 0x4" ::: "memory");     // current 4 loads done
        } else {
            asm volatile("s_wait_asynccnt 0x0" ::: "memory");
        }
        __syncthreads();   // all waves' staging complete

        const int k0 = kb * 32;
        const __bf16* ldsK = &KV[cur][0];
        const __bf16* ldsV = &KV[cur][2048];

        // S = Q K^T from LDS K (B-frag: element e -> hd = kf*32 + e + 16*half)
        const __bf16* kp0 = ldsK + lm * HD + half * 16;
        const __bf16* kp1 = ldsK + (16 + lm) * HD + half * 16;
        v8f S0 = (v8f){}, S1 = (v8f){};
        S0 = wmma_bf16(A0, load_bfrag(kp0),      S0);
        S0 = wmma_bf16(A1, load_bfrag(kp0 + 32), S0);
        S1 = wmma_bf16(A0, load_bfrag(kp1),      S1);
        S1 = wmma_bf16(A1, load_bfrag(kp1 + 32), S1);

        const int key0 = k0 + lm;
        const int key1 = k0 + 16 + lm;
        const int p0   = pad[b * SEQ + key0];
        const int p1   = pad[b * SEQ + key1];

        // online softmax per C-layout row; reductions across 16-lane halves
#pragma unroll
        for (int r = 0; r < 8; ++r) {
            const int qrow = qt * 16 + r + half * 8;
            float v0 = (key0 <= qrow && p0) ? S0[r] * SCL : -1e9f;
            float v1 = (key1 <= qrow && p1) ? S1[r] * SCL : -1e9f;
            float rm = fmaxf(v0, v1);
            rm = fmaxf(rm, __shfl_xor(rm, 1, 32));
            rm = fmaxf(rm, __shfl_xor(rm, 2, 32));
            rm = fmaxf(rm, __shfl_xor(rm, 4, 32));
            rm = fmaxf(rm, __shfl_xor(rm, 8, 32));
            const float mnew  = fmaxf(mrow[r], rm);
            const float alpha = exp2f(mrow[r] - mnew);
            const float e0 = exp2f(v0 - mnew);
            const float e1 = exp2f(v1 - mnew);
            float rs = e0 + e1;
            rs += __shfl_xor(rs, 1, 32);
            rs += __shfl_xor(rs, 2, 32);
            rs += __shfl_xor(rs, 4, 32);
            rs += __shfl_xor(rs, 8, 32);
            lrow[r] = lrow[r] * alpha + rs;
            mrow[r] = mnew;
            o0[r] *= alpha; o1[r] *= alpha; o2[r] *= alpha; o3[r] *= alpha;
            Pl[wid][r + half * 8][lm]      = (__bf16)e0;   // LDS ops are in-order per wave
            Pl[wid][r + half * 8][lm + 16] = (__bf16)e1;
        }

        // P A-fragment: row m = lm, element e -> key = 16*half + e
        const __bf16* pp = &Pl[wid][lm][half * 16];
        v16bf Pf = bfcat(*(const v8bf*)pp, *(const v8bf*)(pp + 8));

        // O += P V from LDS V (row n = hd, 32 keys per row)
        const __bf16* vb = ldsV + lm * 32 + half * 16;
        o0 = wmma_bf16(Pf, load_bfrag(vb),            o0);
        o1 = wmma_bf16(Pf, load_bfrag(vb + 16 * 32),  o1);
        o2 = wmma_bf16(Pf, load_bfrag(vb + 32 * 32),  o2);
        o3 = wmma_bf16(Pf, load_bfrag(vb + 48 * 32),  o3);

        __syncthreads();   // all waves done reading KV[cur] before it is restaged
    }

    // finalize: divide by row sums, store fp32 [B][T][64]
#pragma unroll
    for (int r = 0; r < 8; ++r) {
        const float inv = 1.0f / lrow[r];
        const size_t orow = ((size_t)(b * SEQ + qt * 16 + r + half * 8)) * HD;
        out[orow + lm]      = o0[r] * inv;
        out[orow + 16 + lm] = o1[r] * inv;
        out[orow + 32 + lm] = o2[r] * inv;
        out[orow + 48 + lm] = o3[r] * inv;
    }
}

// ---------------------------------------------------------------------------
// Launch
// ---------------------------------------------------------------------------
extern "C" void kernel_launch(void* const* d_in, const int* in_sizes, int n_in,
                              void* d_out, int out_size, void* d_ws, size_t ws_size,
                              hipStream_t stream) {
    const float* x   = (const float*)d_in[0];
    const float* Wq  = (const float*)d_in[1];
    const float* Wk  = (const float*)d_in[2];
    const float* Wv  = (const float*)d_in[3];
    const int*   pad = (const int*)d_in[4];
    float*       out = (float*)d_out;

    char* ws = (char*)d_ws;
    __bf16* wt = (__bf16*)(ws);                    // 384 KB
    __bf16* q  = (__bf16*)(ws + (512u << 10));     // 2 MB
    __bf16* k  = (__bf16*)(ws + (512u << 10) + (2u << 20));
    __bf16* vt = (__bf16*)(ws + (512u << 10) + (4u << 20));

    k_convw<<<dim3((3 * HD * EMB + 255) / 256), dim3(256), 0, stream>>>(Wq, Wk, Wv, wt);
    k_proj <<<dim3((BATCH * SEQ) / 16), dim3(32), 0, stream>>>(x, wt, q, k, vt);
    k_attn <<<dim3(BATCH * (SEQ / 16) / 4), dim3(128), 0, stream>>>(q, k, vt, pad, out);
}
